// MyRNNLowAPI_79379585564724
// MI455X (gfx1250) — matrix operand order repaired and checked by reading, not compiled, not run on hardware
//
#include <hip/hip_runtime.h>
#include <hip/hip_bf16.h>

// ---------------- problem constants ----------------
#define B_  2048
#define T_  80
#define E_  100
#define V_  10000
#define U_  512
#define KX  128            // E padded up to multiple of 32 for WMMA K
#define BM  128            // workgroup M tile
#define BN  64             // workgroup N tile
#define LDA 40             // LDS row stride (bf16 elems) for A tile, 80B (16B mult, conflict-skewed)
#define LDB 72             // LDS row stride for B tile, 144B

typedef __bf16 bf16x8  __attribute__((ext_vector_type(8)));
typedef __bf16 bf16x16 __attribute__((ext_vector_type(16)));
typedef float  f32x8   __attribute__((ext_vector_type(8)));

// Build a 16-element bf16 fragment from two 16B halves in LDS (ds_load_b128 x2).
__device__ __forceinline__ bf16x16 frag2(const __bf16* p0, const __bf16* p1) {
    bf16x8 lo = *(const bf16x8*)p0;
    bf16x8 hi = *(const bf16x8*)p1;
    return __builtin_shufflevector(lo, hi, 0,1,2,3,4,5,6,7,8,9,10,11,12,13,14,15);
}

// Wait for all outstanding async (direct-to-LDS) transfers of this wave.
__device__ __forceinline__ void wait_async0() {
#if __has_builtin(__builtin_amdgcn_s_wait_asynccnt)
    __builtin_amdgcn_s_wait_asynccnt(0);
#else
    asm volatile("s_wait_asynccnt 0x0" ::: "memory");
#endif
}

// One b128 global -> LDS async copy (no VGPR middleman, ASYNCcnt-tracked).
__device__ __forceinline__ void async_b128(unsigned ldsOff, unsigned long long gaddr) {
    asm volatile("global_load_async_to_lds_b128 %0, %1, off"
                 :: "v"(ldsOff), "v"(gaddr) : "memory");
}

// ---------------- weight conversion (once per launch) ----------------
// Wx fp32 [E,U] -> bf16 [2][KX][U] zero padded in K.
__global__ __launch_bounds__(256) void prep_wx_kernel(const float* __restrict__ wx0,
                                                      const float* __restrict__ wx1,
                                                      unsigned short* __restrict__ outRaw) {
    __bf16* out = (__bf16*)outRaw;
    int i = blockIdx.x * blockDim.x + threadIdx.x;
    if (i >= 2 * KX * U_) return;
    int c   = i / (KX * U_);
    int rem = i % (KX * U_);
    int k   = rem / U_;
    int n   = rem % U_;
    const float* src = c ? wx1 : wx0;
    float v = (k < E_) ? src[k * U_ + n] : 0.0f;
    out[i] = (__bf16)v;
}

// Wh fp32 [U,U] -> bf16 [2][U][U]
__global__ __launch_bounds__(256) void prep_wh_kernel(const float* __restrict__ wh0,
                                                      const float* __restrict__ wh1,
                                                      unsigned short* __restrict__ outRaw) {
    __bf16* out = (__bf16*)outRaw;
    int i = blockIdx.x * blockDim.x + threadIdx.x;
    if (i >= 2 * U_ * U_) return;
    int c   = i / (U_ * U_);
    int rem = i % (U_ * U_);
    const float* src = c ? wh1 : wh0;
    out[i] = (__bf16)src[rem];
}

// ---------------- one recurrent step: h_new = tanh(x_e@Wx + h@Wh + b) ----------------
// grid = (B/BM, U/BN, 2 cells), block = 256 threads (8 waves)
__global__ __launch_bounds__(256) void rnn_step_kernel(
    const int*            __restrict__ x,        // [B,T] tokens
    const float*          __restrict__ emb,      // [V,E] fp32
    const unsigned short* __restrict__ wxAllRaw, // bf16 [2][KX][U]
    const unsigned short* __restrict__ whAllRaw, // bf16 [2][U][U]
    const float*          __restrict__ bias0,
    const float*          __restrict__ bias1,
    const unsigned short* __restrict__ hinRaw,   // bf16 base (buf applied), +cell stride inside
    unsigned short*       __restrict__ houtRaw,
    int t)
{
    __shared__ __align__(16) __bf16 sA[2][BM * LDA];
    __shared__ __align__(16) __bf16 sB[2][32 * LDB];

    const int tid    = threadIdx.x;
    const int lane   = tid & 31;
    const int wave   = tid >> 5;
    const int wm     = wave & 1;   // 0..1 -> M half (64 rows)
    const int wn     = wave >> 1;  // 0..3 -> N quarter (16 cols)
    const int cell   = blockIdx.z;
    const int blockM = blockIdx.x * BM;
    const int blockN = blockIdx.y * BN;

    const size_t HC = (size_t)2 * B_ * U_;   // cell stride (elements) in H buffer
    const __bf16* wx   = (const __bf16*)wxAllRaw + (size_t)cell * KX * U_;
    const __bf16* wh   = (const __bf16*)whAllRaw + (size_t)cell * U_ * U_;
    const __bf16* hin  = (const __bf16*)hinRaw  + (size_t)cell * HC;
    __bf16*       hout = (__bf16*)houtRaw       + (size_t)cell * HC;
    const float*  bias = cell ? bias1 : bias0;

    f32x8 acc0 = {}, acc1 = {}, acc2 = {}, acc3 = {};

    // A-stage indices: each thread fills 16 contiguous K of one row
    const int arow = tid >> 1;         // 0..127
    const int ak0  = (tid & 1) * 16;   // 0 or 16
    // B-stage indices: each thread fills 8 contiguous N of one K row
    const int bk  = tid >> 3;          // 0..31
    const int bn0 = (tid & 7) * 8;     // 0..56

    const int    tok  = x[(size_t)(blockM + arow) * T_ + t];
    const float* erow = emb + (size_t)tok * E_;

    // fragment indices
    const int lm = lane & 15;
    const int kb = (lane >> 4) << 3;   // K base 0 or 8 per ISA A layout

    // Stage K-chunk `chunk` into LDS buffer `buf`.
    // chunks 0..3: embedded-input part (fp32 gather+convert, ds_store path)
    // chunks 4..19: recurrent part (bf16 h state, async direct-to-LDS)
    // B tile is always bf16 weights -> always async direct-to-LDS.
    auto stage = [&](int chunk, int buf) {
        __bf16* dA = &sA[buf][0];
        __bf16* dB = &sB[buf][0];
        if (chunk < 4) {
            int kk = chunk * 32 + ak0;
            #pragma unroll
            for (int i = 0; i < 16; ++i) {
                float v = (kk + i < E_) ? erow[kk + i] : 0.0f;
                dA[arow * LDA + ak0 + i] = (__bf16)v;
            }
        } else {
            int kk = (chunk - 4) * 32 + ak0;
            unsigned long long g =
                (unsigned long long)(hin + (size_t)(blockM + arow) * U_ + kk);
            unsigned l = (unsigned)(size_t)&dA[arow * LDA + ak0];
            async_b128(l,       g);
            async_b128(l + 16u, g + 16ull);
        }
        const __bf16* wsrc = (chunk < 4)
            ? (wx + (size_t)(chunk * 32 + bk) * U_ + blockN + bn0)
            : (wh + (size_t)((chunk - 4) * 32 + bk) * U_ + blockN + bn0);
        async_b128((unsigned)(size_t)&dB[bk * LDB + bn0],
                   (unsigned long long)wsrc);
    };

    // prologue: stage chunk 0 into buffer 0
    stage(0, 0);
    wait_async0();
    __syncthreads();

    // K loop, double-buffered: stage chunk+1 while computing chunk
    for (int chunk = 0; chunk < 20; ++chunk) {
        const int cur = chunk & 1;
        if (chunk + 1 < 20) stage(chunk + 1, cur ^ 1);

        // ---- compute: 4 WMMAs per wave (64M x 16N sub-tile) ----
        const __bf16* cA = &sA[cur][0];
        const __bf16* cB = &sB[cur][0];
        const __bf16* bp = &cB[kb * LDB + wn * 16 + lm];   // lane = N col, elems = K (table order)
        bf16x16 bfrag;
        #pragma unroll
        for (int s = 0; s < 16; ++s) bfrag[s] = bp[(s < 8 ? s : s + 8) * LDB];
        const __bf16* abase = &cA[(size_t)(wm * 64 + lm) * LDA + kb];
        bf16x16 a0 = frag2(abase,            abase + 16);
        bf16x16 a1 = frag2(abase + 16 * LDA, abase + 16 * LDA + 16);
        bf16x16 a2 = frag2(abase + 32 * LDA, abase + 32 * LDA + 16);
        bf16x16 a3 = frag2(abase + 48 * LDA, abase + 48 * LDA + 16);
        acc0 = __builtin_amdgcn_wmma_f32_16x16x32_bf16(false, a0, false, bfrag, (short)0, acc0, false, false);
        acc1 = __builtin_amdgcn_wmma_f32_16x16x32_bf16(false, a1, false, bfrag, (short)0, acc1, false, false);
        acc2 = __builtin_amdgcn_wmma_f32_16x16x32_bf16(false, a2, false, bfrag, (short)0, acc2, false, false);
        acc3 = __builtin_amdgcn_wmma_f32_16x16x32_bf16(false, a3, false, bfrag, (short)0, acc3, false, false);

        // single barrier point: next tile landed AND current tile fully read
        wait_async0();
        __syncthreads();
    }

    // ---- epilogue: bias + tanh, store bf16 h_new ----
    const int col  = blockN + wn * 16 + lm;
    const float bv = bias[col];
    const int rsub = (lane >> 4) << 3;   // lanes 16-31 hold M rows 8..15 per C layout
    #pragma unroll
    for (int m = 0; m < 4; ++m) {
        f32x8 a = (m == 0) ? acc0 : (m == 1) ? acc1 : (m == 2) ? acc2 : acc3;
        int rowBase = blockM + wm * 64 + m * 16 + rsub;
        #pragma unroll
        for (int r = 0; r < 8; ++r) {
            float v = a[r] + bv;
            hout[(size_t)(rowBase + r) * U_ + col] = (__bf16)tanhf(v);
        }
    }
}

// ---------------- output head: sigmoid(h1 @ Wout + bout) ----------------
// one wave per batch row, 8 rows per 256-thread block
__global__ __launch_bounds__(256) void rnn_out_kernel(const unsigned short* __restrict__ h1Raw,
                                                      const float* __restrict__ wout,
                                                      const float* __restrict__ bout,
                                                      float* __restrict__ out) {
    const int wave = threadIdx.x >> 5;
    const int lane = threadIdx.x & 31;
    const int b    = blockIdx.x * 8 + wave;
    if (b >= B_) return;
    const __bf16* hr = (const __bf16*)h1Raw + (size_t)b * U_;
    float s = 0.0f;
    #pragma unroll
    for (int i = 0; i < U_ / 32; ++i) {
        int u = lane + 32 * i;
        s += (float)hr[u] * wout[u];
    }
    #pragma unroll
    for (int off = 16; off > 0; off >>= 1)
        s += __shfl_down(s, off, 32);
    if (lane == 0)
        out[b] = 1.0f / (1.0f + expf(-(s + bout[0])));
}

// ---------------- host launcher ----------------
extern "C" void kernel_launch(void* const* d_in, const int* in_sizes, int n_in,
                              void* d_out, int out_size, void* d_ws, size_t ws_size,
                              hipStream_t stream) {
    const int*   x    = (const int*)d_in[0];
    const float* emb  = (const float*)d_in[1];
    const float* Wx0  = (const float*)d_in[2];
    const float* Wh0  = (const float*)d_in[3];
    const float* b0   = (const float*)d_in[4];
    const float* Wx1  = (const float*)d_in[5];
    const float* Wh1  = (const float*)d_in[6];
    const float* b1   = (const float*)d_in[7];
    const float* Wout = (const float*)d_in[8];
    const float* bout = (const float*)d_in[9];
    float* out = (float*)d_out;

    // workspace layout (bf16 elements):
    //   wx [2][KX][U]  : 2*128*512           = 131072
    //   wh [2][U][U]   : 2*512*512           = 524288
    //   H  [2 cells][2 bufs][B][U]           = 4194304
    unsigned short* wx = (unsigned short*)d_ws;
    unsigned short* wh = wx + (size_t)2 * KX * U_;
    unsigned short* H  = wh + (size_t)2 * U_ * U_;
    const size_t HB = (size_t)B_ * U_;     // buffer stride (elements)
    const size_t HC = 2 * HB;              // cell stride (elements)

    // zero both ping-pong h buffers for both cells (t=0 reads buf0 zeros)
    hipMemsetAsync(H, 0, (size_t)2 * HC * sizeof(unsigned short), stream);

    prep_wx_kernel<<<(2 * KX * U_ + 255) / 256, 256, 0, stream>>>(Wx0, Wx1, wx);
    prep_wh_kernel<<<(2 * U_ * U_ + 255) / 256, 256, 0, stream>>>(Wh0, Wh1, wh);

    dim3 grid(B_ / BM, U_ / BN, 2);
    for (int t = 0; t < T_; ++t) {
        unsigned short* hin  = H + (size_t)(t & 1) * HB;
        unsigned short* hout = H + (size_t)((t + 1) & 1) * HB;
        rnn_step_kernel<<<grid, 256, 0, stream>>>(x, emb, wx, wh, b0, b1, hin, hout, t);
    }

    // final h1 lives at cell 1, buffer (T%2)==0
    rnn_out_kernel<<<B_ / 8, 256, 0, stream>>>(H + HC, Wout, bout, out);
}
